// Attention_3616362463521
// MI455X (gfx1250) — matrix operand order repaired and checked by reading, not compiled, not run on hardware
//
#include <hip/hip_runtime.h>
#include <hip/hip_bf16.h>

#define B_ 32
#define S_ 2048
#define H_ 1024

typedef __attribute__((ext_vector_type(16))) __bf16 bf16x16;
typedef __attribute__((ext_vector_type(8)))  __bf16 bf16x8;
typedef __attribute__((ext_vector_type(8)))  float  f32x8;
typedef unsigned int v4u __attribute__((ext_vector_type(4)));
typedef int          v8i __attribute__((ext_vector_type(8)));
typedef int          i32x4_ __attribute__((ext_vector_type(4)));

typedef __attribute__((address_space(1))) i32x4_ g_i32x4;   // global 128-bit chunk
typedef __attribute__((address_space(3))) i32x4_ l_i32x4;   // LDS 128-bit chunk

// ---- async global->LDS copy (per-lane b128), guarded with sync fallback ----
#if defined(__has_builtin)
#  if __has_builtin(__builtin_amdgcn_global_load_async_to_lds_b128)
#    define ASYNC_LDS 1
#  endif
#endif
#ifndef ASYNC_LDS
#  define ASYNC_LDS 0
#endif

__device__ __forceinline__ void copy16_g2l(const void* g, void* l) {
#if ASYNC_LDS
    __builtin_amdgcn_global_load_async_to_lds_b128(
        (g_i32x4*)(unsigned long long)g,
        (l_i32x4*)(unsigned)(unsigned long long)l, 0, 0);
#else
    *(uint4*)l = *(const uint4*)g;
#endif
}

__device__ __forceinline__ void wait_async_lds() {
#if ASYNC_LDS
    asm volatile("s_wait_asynccnt 0x0" ::: "memory");
#endif
}

// LDS byte offset of a generic shared pointer (flat LDS addr low 32 bits = DS address)
__device__ __forceinline__ unsigned ldsoff(const void* p) {
    return (unsigned)(unsigned long long)p;
}

// ---- Tensor Data Mover: load a 32(cols, bf16) x 128(rows) tile, row stride 1024
// elements, into LDS with 16B padding after every 64B row (pad_interval=16 DW -> code 3,
// pad_amount=4 DW -> code 3). D# per CDNA5 ISA 8.3-8.5; issued wave-uniform.
__device__ __forceinline__ void tdm_load_we(const __bf16* gsrc, unsigned lds_byte_off) {
    unsigned long long ga = (unsigned long long)gsrc;
    v4u g0;
    g0[0] = 1u;                                        // count=1, user descriptor
    g0[1] = lds_byte_off;                              // lds_addr
    g0[2] = (unsigned)ga;                              // global_addr[31:0]
    g0[3] = (unsigned)((ga >> 32) & 0x01FFFFFFull) | (2u << 30); // addr[56:32] | type=2
    v8i g1;
    g1[0] = (1 << 16) | (1 << 20) | (3 << 22) | (3 << 25); // data_size=2B, pad_en, intvl, amt
    g1[1] = (int)(1024u << 16);                        // tensor_dim0[15:0] = 1024
    g1[2] = (int)(1024u << 16);                        // dim0 hi=0 | tensor_dim1[15:0]=1024
    g1[3] = (int)(32u << 16);                          // dim1 hi=0 | tile_dim0=32
    g1[4] = 128;                                       // tile_dim1=128, tile_dim2=0
    g1[5] = 1024;                                      // tensor_dim0_stride[31:0]
    g1[6] = 0;
    g1[7] = 0;
    asm volatile("tensor_load_to_lds %0, %1" :: "s"(g0), "s"(g1) : "memory");
}

// ---------------- kernel 0: split "we" half of attn_w into bf16 hi/lo planes ----------------
__global__ __launch_bounds__(256) void k_convert_we(const float* __restrict__ attn_w,
                                                    __bf16* __restrict__ we_hi,
                                                    __bf16* __restrict__ we_lo) {
    int idx = blockIdx.x * 256 + threadIdx.x;
    int k = idx >> 10, h = idx & 1023;
    float x = attn_w[(size_t)k * (2 * H_) + H_ + h];
    __bf16 hi = (__bf16)x;
    we_hi[idx] = hi;
    we_lo[idx] = (__bf16)(x - (float)hi);
}

// ---------------- kernel 1: q[b,k] = hidden[b]·wh[k] + bias[k] ----------------
__global__ __launch_bounds__(256) void k_query(const float* __restrict__ hidden,
                                               const float* __restrict__ attn_w,
                                               const float* __restrict__ attn_b,
                                               float* __restrict__ q) {
    int b = blockIdx.y;
    int k = blockIdx.x * 256 + threadIdx.x;
    __shared__ __align__(16) float sh[H_];
    for (int i = threadIdx.x; i < H_; i += 256) sh[i] = hidden[b * H_ + i];
    __syncthreads();
    const float4* w4 = (const float4*)(attn_w + (size_t)k * (2 * H_));
    const float4* h4 = (const float4*)sh;
    float acc = attn_b[k];
#pragma unroll 4
    for (int i = 0; i < H_ / 4; ++i) {
        float4 w = w4[i], hh = h4[i];
        acc += w.x * hh.x + w.y * hh.y + w.z * hh.z + w.w * hh.w;
    }
    q[b * H_ + k] = acc;
}

// ---------------- kernel 2: fused GEMM + tanh + v-dot -> scores[b,s] ----------------
// grid (S/64, B), block 256 (8 waves). 64 s-rows per WG; 8 k-passes of 128 cols.
// Double-buffered LDS; enc staged via async b128 copies, we via TDM tensor loads.
// 2-term bf16 split: C += Ahi*Bhi + Ahi*Blo + Alo*Bhi (f32 accumulate).
__global__ __launch_bounds__(256, 2) void k_scores(const float* __restrict__ enc,
                                                   const __bf16* __restrict__ we_hi,
                                                   const __bf16* __restrict__ we_lo,
                                                   const float* __restrict__ q,
                                                   const float* __restrict__ v_w,
                                                   float* __restrict__ scores) {
    const int b  = blockIdx.y;
    const int s0 = blockIdx.x * 64;
    const int tid = threadIdx.x;

    __shared__ __align__(16) float  sEncF[2][64][36];   // 32 f32 cols + 4 pad
    __shared__ __align__(16) __bf16 sWeHi[2][128][40];  // 32 bf16 cols + 8 pad
    __shared__ __align__(16) __bf16 sWeLo[2][128][40];
    __shared__ float sScore[64];

    if (tid < 64) sScore[tid] = 0.0f;

    const int lane  = tid & 31;
    const int wv    = tid >> 5;
    const int sSub  = wv & 3;       // s rows sSub*16 .. +15
    const int kHalf = wv >> 2;      // k cols kHalf*64 .. +63
    const int n     = lane & 15;
    const int g     = lane >> 4;

    auto stage = [&](int kp, int ho, int buf) {
#pragma unroll
        for (int j = 0; j < 2; ++j) {
            int idx = tid + j * 256;                 // 0..511
            int r = idx >> 3, c4 = idx & 7;
            copy16_g2l(enc + ((size_t)(b * S_ + s0 + r)) * H_ + ho + c4 * 4,
                       &sEncF[buf][r][c4 * 4]);
        }
        if (wv == 0) {
            tdm_load_we(we_hi + (size_t)(kp * 128) * H_ + ho, ldsoff(&sWeHi[buf][0][0]));
            tdm_load_we(we_lo + (size_t)(kp * 128) * H_ + ho, ldsoff(&sWeLo[buf][0][0]));
        }
    };
    auto wait_stage = [&]() {
        wait_async_lds();
        if (wv == 0) __builtin_amdgcn_s_wait_tensorcnt(0);
        __syncthreads();
    };
    auto compute = [&](int buf, f32x8* acc) {
        const int ar = sSub * 16 + n;
        float fa[16];
        *(float4*)&fa[0]  = *(const float4*)&sEncF[buf][ar][g * 16];
        *(float4*)&fa[4]  = *(const float4*)&sEncF[buf][ar][g * 16 + 4];
        *(float4*)&fa[8]  = *(const float4*)&sEncF[buf][ar][g * 16 + 8];
        *(float4*)&fa[12] = *(const float4*)&sEncF[buf][ar][g * 16 + 12];
        bf16x16 Ahi, Alo;
#pragma unroll
        for (int i = 0; i < 16; ++i) {
            __bf16 h = (__bf16)fa[i];
            Ahi[i] = h;
            Alo[i] = (__bf16)(fa[i] - (float)h);
        }
#pragma unroll
        for (int kt = 0; kt < 4; ++kt) {
            const int kr = kHalf * 64 + kt * 16 + n;
            bf16x8 b0 = *(const bf16x8*)&sWeHi[buf][kr][g * 16];
            bf16x8 b1 = *(const bf16x8*)&sWeHi[buf][kr][g * 16 + 8];
            bf16x8 b2 = *(const bf16x8*)&sWeLo[buf][kr][g * 16];
            bf16x8 b3 = *(const bf16x8*)&sWeLo[buf][kr][g * 16 + 8];
            bf16x16 Bhi = __builtin_shufflevector(b0, b1, 0,1,2,3,4,5,6,7,8,9,10,11,12,13,14,15);
            bf16x16 Blo = __builtin_shufflevector(b2, b3, 0,1,2,3,4,5,6,7,8,9,10,11,12,13,14,15);
            acc[kt] = __builtin_amdgcn_wmma_f32_16x16x32_bf16(
                false, Ahi, false, Bhi, (short)0, acc[kt], false, false);
            acc[kt] = __builtin_amdgcn_wmma_f32_16x16x32_bf16(
                false, Ahi, false, Blo, (short)0, acc[kt], false, false);
            acc[kt] = __builtin_amdgcn_wmma_f32_16x16x32_bf16(
                false, Alo, false, Bhi, (short)0, acc[kt], false, false);
        }
    };

    float sacc[8];
#pragma unroll
    for (int r = 0; r < 8; ++r) sacc[r] = 0.0f;

    for (int kp = 0; kp < 8; ++kp) {
        f32x8 acc[4];
#pragma unroll
        for (int kt = 0; kt < 4; ++kt) {
            float qv = q[b * H_ + kp * 128 + kHalf * 64 + kt * 16 + n];
#pragma unroll
            for (int r = 0; r < 8; ++r) acc[kt][r] = qv;
        }

        __syncthreads();
        stage(kp, 0, 0);
        wait_stage();
        for (int i = 0; i < 32; ++i) {
            int buf = i & 1;
            if (i + 1 < 32) stage(kp, (i + 1) * 32, buf ^ 1);  // overlap next stage
            compute(buf, acc);
            wait_stage();
        }

        // epilogue: tanh(energy)*v accumulated per lane
#pragma unroll
        for (int kt = 0; kt < 4; ++kt) {
            float vw = v_w[kp * 128 + kHalf * 64 + kt * 16 + n];
#pragma unroll
            for (int r = 0; r < 8; ++r)
                sacc[r] += tanhf(acc[kt][r]) * vw;
        }
    }

    // reduce over 16 N-lanes per half-group, then LDS-atomic into row scores
#pragma unroll
    for (int r = 0; r < 8; ++r) {
        float v = sacc[r];
        v += __shfl_xor(v, 1, 32);
        v += __shfl_xor(v, 2, 32);
        v += __shfl_xor(v, 4, 32);
        v += __shfl_xor(v, 8, 32);
        if (n == 0) atomicAdd(&sScore[sSub * 16 + g * 8 + r], v);
    }
    __syncthreads();
    if (tid < 64) scores[b * S_ + s0 + tid] = sScore[tid];
}

// ---------------- kernel 3: softmax over S per batch ----------------
__global__ __launch_bounds__(256) void k_softmax(const float* __restrict__ scores,
                                                 float* __restrict__ weights) {
    int b = blockIdx.x, tid = threadIdx.x;
    __shared__ float red[256];
    float v[8];
    float mx = -3.4e38f;
#pragma unroll
    for (int i = 0; i < 8; ++i) {
        v[i] = scores[b * S_ + tid + i * 256];
        mx = fmaxf(mx, v[i]);
    }
    red[tid] = mx; __syncthreads();
    for (int o = 128; o > 0; o >>= 1) {
        if (tid < o) red[tid] = fmaxf(red[tid], red[tid + o]);
        __syncthreads();
    }
    mx = red[0]; __syncthreads();
    float sum = 0.f;
#pragma unroll
    for (int i = 0; i < 8; ++i) { v[i] = __expf(v[i] - mx); sum += v[i]; }
    red[tid] = sum; __syncthreads();
    for (int o = 128; o > 0; o >>= 1) {
        if (tid < o) red[tid] += red[tid + o];
        __syncthreads();
    }
    float inv = 1.0f / red[0];
#pragma unroll
    for (int i = 0; i < 8; ++i) weights[b * S_ + tid + i * 256] = v[i] * inv;
}

// ---------------- kernel 4: context partial sums over S chunks ----------------
__global__ __launch_bounds__(256) void k_ctx_partial(const float* __restrict__ enc,
                                                     const float* __restrict__ weights,
                                                     float* __restrict__ part) {
    int b = blockIdx.y, c = blockIdx.x, tid = threadIdx.x;
    __shared__ float sw[128];
    if (tid < 128) sw[tid] = weights[b * S_ + c * 128 + tid];
    __syncthreads();
    float4 acc = {0.f, 0.f, 0.f, 0.f};
    for (int s = 0; s < 128; ++s) {
        const float4* ep = (const float4*)(enc + ((size_t)(b * S_ + c * 128 + s)) * H_);
        float4 ev = ep[tid];
        float w = sw[s];
        acc.x += w * ev.x; acc.y += w * ev.y; acc.z += w * ev.z; acc.w += w * ev.w;
    }
    ((float4*)part)[((size_t)(c * B_ + b) * H_) / 4 + tid] = acc;
}

// ---------------- kernel 5: reduce partials -> context ----------------
__global__ __launch_bounds__(256) void k_ctx_reduce(const float* __restrict__ part,
                                                    float* __restrict__ ctx) {
    int b = blockIdx.x, tid = threadIdx.x;
    float4 s = {0.f, 0.f, 0.f, 0.f};
#pragma unroll
    for (int c = 0; c < 16; ++c) {
        float4 p = ((const float4*)part)[((size_t)(c * B_ + b) * H_) / 4 + tid];
        s.x += p.x; s.y += p.y; s.z += p.z; s.w += p.w;
    }
    ((float4*)ctx)[((size_t)b * H_) / 4 + tid] = s;
}

extern "C" void kernel_launch(void* const* d_in, const int* in_sizes, int n_in,
                              void* d_out, int out_size, void* d_ws, size_t ws_size,
                              hipStream_t stream) {
    const float* hidden = (const float*)d_in[0];   // (1,B,H)
    const float* enc    = (const float*)d_in[1];   // (B,S,H)
    const float* attn_w = (const float*)d_in[2];   // (H,2H)
    const float* attn_b = (const float*)d_in[3];   // (H,)
    const float* v_w    = (const float*)d_in[4];   // (1,H)

    float* out_ctx = (float*)d_out;                // B*H
    float* out_w   = (float*)d_out + B_ * H_;      // B*S

    char* ws = (char*)d_ws;
    float*  q      = (float*)(ws);                               // 128 KB
    float*  scores = (float*)(ws + 131072);                      // 256 KB
    __bf16* we_hi  = (__bf16*)(ws + 393216);                     // 2 MB
    __bf16* we_lo  = (__bf16*)(ws + 393216 + 2097152);           // 2 MB
    float*  part   = (float*)(ws + 393216 + 2 * 2097152);        // 2 MB

    k_convert_we <<<dim3((H_ * H_) / 256), 256, 0, stream>>>(attn_w, we_hi, we_lo);
    k_query      <<<dim3(H_ / 256, B_),    256, 0, stream>>>(hidden, attn_w, attn_b, q);
    k_scores     <<<dim3(S_ / 64, B_),     256, 0, stream>>>(enc, we_hi, we_lo, q, v_w, scores);
    k_softmax    <<<dim3(B_),              256, 0, stream>>>(scores, out_w);
    k_ctx_partial<<<dim3(16, B_),          256, 0, stream>>>(enc, out_w, part);
    k_ctx_reduce <<<dim3(B_),              256, 0, stream>>>(part, out_ctx);
}
